// TemporalDecoderLSTMConf_5549097746441
// MI455X (gfx1250) — matrix operand order repaired and checked by reading, not compiled, not run on hardware
//
#include <hip/hip_runtime.h>

typedef _Float16 h16;
typedef __attribute__((ext_vector_type(16))) _Float16 v16h;
typedef __attribute__((ext_vector_type(8)))  _Float16 v8h;
typedef __attribute__((ext_vector_type(8)))  float    v8f;

#define NPED  65536
#define HDIM  64
#define EMB   32
#define DDIM  120
#define OUTD  9
#define SEQL  30
#define NW    8
#define TPB   (NW*32)

// ---- fragment ids (512 halfs each) ----
#define FSH(nt,kt) ((nt)*4+(kt))      // 0..7  shared-part SE B' (init only)
#define FTL(nt,kt) (8+(nt)*2+(kt))    // 8..11 tail SE B' (K layout l*16+i, row48=-G)
#define FIH(ut)    (12+(ut))          // 12..27 W_ih
#define FHH(ut,kt) (28+(ut)*2+(kt))   // 28..59 W_hh
#define FHP(kt)    (60+(kt))          // 60..61 W_hp (ln2_g folded, cols>=9 zero)
#define NFRAG   62

// ---- LDS layout (bytes) ----
#define OFF_B   0
#define OFF_F   (NFRAG*1024)                 // 63488
#define F_GBIAS 0                            // 256: b_ih+b_hh
#define F_G     256                          // 32 : sum_k ln1_g*W_se
#define F_BC    288                          // 32 : b_se + sum_k ln1_b*W_se
#define F_BC2   320                          // 16 : b_hp + sum_k ln2_b*W_hp
#define NFLOATS 336
#define OFF_TT  (OFF_F + NFLOATS*4)          // 64832 ; 16 peds x 3 slots x 16 halfs
#define OFF_HS  (OFF_TT + NW*16*48*2)        // +12288
#define OFF_DS  (OFF_HS + NW*16*HDIM*2)      // +16384
#define SMEM_BYTES (OFF_DS + NW*16*EMB*2)    // +8192

#if __has_builtin(__builtin_amdgcn_tanhf)
__device__ __forceinline__ float tanh_fast(float x) { return __builtin_amdgcn_tanhf(x); }
#else
__device__ __forceinline__ float tanh_fast(float x) { return 2.f/(1.f+__expf(-2.f*x)) - 1.f; }
#endif
// sigmoid(x) = 0.5*tanh(0.5*x) + 0.5 ; callers pre-fold the 0.5 scale into an fma

// B-fragment: lane reads 32 contiguous bytes (two ds_load_b128)
__device__ __forceinline__ v16h ld_bfrag(const h16* Bh, int f, int lane) {
  const h16* p = Bh + f*512 + lane*16;
  v8h lo = *(const v8h*)p;
  v8h hi = *(const v8h*)(p+8);
  v16h a;
#pragma unroll
  for (int j=0;j<8;++j){ a[j]=lo[j]; a[j+8]=hi[j]; }
  return a;
}
// A-fragment from contiguous f16 row: element j = K  kb + (j>=8?16:0) + goff + (j&7)
__device__ __forceinline__ v16h ld_arow(const h16* row, int kb, int goff) {
  v8h lo = *(const v8h*)(row + kb + goff);
  v8h hi = *(const v8h*)(row + kb + 16 + goff);
  v16h a;
#pragma unroll
  for (int j=0;j<8;++j){ a[j]=lo[j]; a[j+8]=hi[j]; }
  return a;
}

#define WMMA(A,B,C) __builtin_amdgcn_wmma_f32_16x16x32_f16(false,(A),false,(B),(short)0,(C),false,false)

__global__ __launch_bounds__(TPB)
void tdlstm_kernel(const float* __restrict__ traj_rel,
                   const float* __restrict__ h0,
                   const float* __restrict__ c0,
                   const float* __restrict__ W_ih,
                   const float* __restrict__ W_hh,
                   const float* __restrict__ b_ih,
                   const float* __restrict__ b_hh,
                   const float* __restrict__ ln1_g,
                   const float* __restrict__ ln1_b,
                   const float* __restrict__ W_se,
                   const float* __restrict__ b_se,
                   const float* __restrict__ ln2_g,
                   const float* __restrict__ ln2_b,
                   const float* __restrict__ W_hp,
                   const float* __restrict__ b_hp,
                   float* __restrict__ out)
{
  extern __shared__ char smem[];
  h16*   BhW = (h16*)(smem + OFF_B);
  float* Fs  = (float*)(smem + OFF_F);

  const int tid  = threadIdx.x;
  const int wave = tid >> 5;
  const int lane = tid & 31;
  const int ln16 = lane & 15;
  const int grp  = lane >> 4;
  const int goff = grp * 8;
  const int pedBase = (blockIdx.x*NW + wave)*16;

  h16* TTw = (h16*)(smem + OFF_TT) + wave*16*48;
  h16* Hsw = (h16*)(smem + OFF_HS) + wave*16*HDIM;
  h16* Dsw = (h16*)(smem + OFF_DS) + wave*16*EMB;

  // ================= phase A: scalar folded constants =================
  for (int i = tid; i < 256; i += TPB) Fs[F_GBIAS + i] = b_ih[i] + b_hh[i];
  if (tid < 32) {               // G[e], Bconst[e]
    float g = 0.f, bc = b_se[tid];
    for (int k = 0; k < DDIM; ++k) {
      float w = W_se[tid*DDIM + k];
      g  += ln1_g[k]*w;
      bc += ln1_b[k]*w;
    }
    Fs[F_G + tid] = g; Fs[F_BC + tid] = bc;
  } else if (tid < 48) {        // Bc2[o]
    int o = tid - 32;
    float bc = 0.f;
    if (o < OUTD) {
      bc = b_hp[o];
      for (int k = 0; k < HDIM; ++k) bc += ln2_b[k]*W_hp[o*HDIM + k];
    }
    Fs[F_BC2 + o] = bc;
  }
  __syncthreads();

  // ================= phase B: f16 B-fragments =================
  for (int idx = tid; idx < NFRAG*512; idx += TPB) {
    int f = idx >> 9, pos = idx & 511, L = pos >> 4, j = pos & 15;
    int n = L & 15;
    int K = ((j >= 8) ? 16 : 0) + ((L >= 16) ? 8 : 0) + (j & 7);
    float v = 0.f;
    if (f < 8) {                        // shared SE B' (tail rows zeroed)
      int nt = f >> 2, kt = f & 3, k = kt*32 + K;
      if (k < DDIM && (k % 40) < 31) v = ln1_g[k] * W_se[(nt*16 + n)*DDIM + k];
    } else if (f < 12) {                // tail SE B', K layout l*16+i, row48=-G
      int ff = f - 8, nt = ff >> 1, kt = ff & 1, kp = kt*32 + K, e = nt*16 + n;
      if (kp < 48) {
        int l = kp >> 4, i = kp & 15;
        if (i < OUTD) { int k = l*40 + 31 + i; v = ln1_g[k] * W_se[e*DDIM + k]; }
      } else if (kp == 48) v = -Fs[F_G + e];
    } else if (f < 28) {                // W_ih: B[e][u]
      int ut = f - 12;
      v = W_ih[(ut*16 + n)*EMB + K];
    } else if (f < 60) {                // W_hh: B[k][u]
      int g2 = f - 28, ut = g2 >> 1, kt = g2 & 1;
      v = W_hh[(ut*16 + n)*HDIM + kt*32 + K];
    } else {                            // W_hp B2 (ln2_g folded)
      int kt = f - 60, k = kt*32 + K;
      if (n < OUTD) v = ln2_g[k] * W_hp[n*HDIM + k];
    }
    BhW[f*512 + L*16 + j] = (h16)v;
  }

  // ================= per-wave state init =================
  for (int idx = lane; idx < 16*48; idx += 32) {     // tails (3 identical slots, padded)
    int p = idx / 48, q_ = idx % 48, i = q_ & 15;
    float v = 0.f;
    if (i < OUTD) { int o = 31 + i; v = traj_rel[((size_t)(o>>1)*NPED + pedBase + p)*2 + (o&1)]; }
    TTw[idx] = (h16)v;
  }
  for (int idx = lane; idx < 16*HDIM; idx += 32) {   // h0
    int p = idx >> 6, u = idx & 63;
    Hsw[idx] = (h16)h0[(size_t)(pedBase + p)*HDIM + u];
  }
  float cf[4][8];
#pragma unroll
  for (int jg = 0; jg < 4; ++jg)
#pragma unroll
    for (int r = 0; r < 8; ++r)
      cf[jg][r] = c0[(size_t)(pedBase + grp*8 + r)*HDIM + jg*16 + ln16];

  // shared-part LN1 running sums (per ped = ln16; identical in both half-waves)
  float sSh = 0.f, qSh = 0.f;
  for (int o = 0; o < 31; ++o) {
    float x = (float)(h16)traj_rel[((size_t)(o>>1)*NPED + pedBase + ln16)*2 + (o&1)];
    sSh += x; qSh += x*x;
  }
  sSh *= 3.f; qSh *= 3.f;

  __syncthreads();

  // C_ped = X_shared @ B'_shared  (tail rows of B' are zero, so feed full X)
  v8f cp0, cp1;
#pragma unroll
  for (int r = 0; r < 8; ++r) { cp0[r] = 0.f; cp1[r] = 0.f; }
#pragma unroll
  for (int kt = 0; kt < 4; ++kt) {
    v16h a;
#pragma unroll
    for (int j = 0; j < 16; ++j) {
      int k = kt*32 + ((j >= 8) ? 16 : 0) + goff + (j & 7);
      h16 x = (h16)0.f;
      if (k < DDIM) { int o = k % 40; x = (h16)traj_rel[((size_t)(o>>1)*NPED + pedBase + ln16)*2 + (o&1)]; }
      a[j] = x;
    }
    cp0 = WMMA(a, ld_bfrag(BhW, FSH(0,kt), lane), cp0);
    cp1 = WMMA(a, ld_bfrag(BhW, FSH(1,kt), lane), cp1);
  }

  const h16* TTp  = TTw + ln16*48;
  const h16* hrow = Hsw + ln16*HDIM;
  const float bc0 = Fs[F_BC + ln16], bc1 = Fs[F_BC + 16 + ln16];
  const float bc2 = Fs[F_BC2 + ln16];

  // hoist gate biases into registers; sigmoid gates pre-scaled by 0.5
  float gbI[4], gbF[4], gbG[4], gbO[4];
#pragma unroll
  for (int jg = 0; jg < 4; ++jg) {
    gbI[jg] = 0.5f*Fs[F_GBIAS + (0*4+jg)*16 + ln16];
    gbF[jg] = 0.5f*Fs[F_GBIAS + (1*4+jg)*16 + ln16];
    gbG[jg] =      Fs[F_GBIAS + (2*4+jg)*16 + ln16];
    gbO[jg] = 0.5f*Fs[F_GBIAS + (3*4+jg)*16 + ln16];
  }

  // ================= recurrent loop =================
  for (int t = 0; t < SEQL; ++t) {
    int r0  = t % 3;
    int pb0 = r0;
    int pb1 = (r0 >= 2) ? 0 : r0 + 1;
    int pb2 = (r0 >= 1) ? r0 - 1 : 2;

    // ---- LN1 stats: shared consts + tail re-sum split across half-waves ----
    float s = 0.f, q = 0.f;
#pragma unroll
    for (int w = 0; w < 3; ++w) {
      v8h v = *(const v8h*)(TTp + grp*24 + w*8);
#pragma unroll
      for (int jj = 0; jj < 8; ++jj) { float x = (float)v[jj]; s += x; q += x*x; }
    }
    s += __shfl_xor(s, 16, 32); q += __shfl_xor(q, 16, 32);
    s += sSh; q += qSh;
    float mu = s * (1.f/120.f);
    float rs = rsqrtf(q * (1.f/120.f) - mu*mu + 1e-5f);

    // ---- SE: d = rs*(C_ped + tail@B'tail - mu*G) + Bconst ; leaky_relu ----
    v8h t0lo = *(const v8h*)(TTp + pb0*16 + goff);
    v8h t0hi = *(const v8h*)(TTp + pb1*16 + goff);
    v8h t1lo = *(const v8h*)(TTp + pb2*16 + goff);
    v16h a0, a1;
#pragma unroll
    for (int j = 0; j < 8; ++j) { a0[j]=t0lo[j]; a0[j+8]=t0hi[j]; a1[j]=t1lo[j]; a1[j+8]=(h16)0.f; }
    a1[8] = (grp == 0) ? (h16)mu : (h16)0.f;   // K=48 row of B holds -G[e]

    v8f d0 = WMMA(a0, ld_bfrag(BhW, FTL(0,0), lane), cp0);
    d0     = WMMA(a1, ld_bfrag(BhW, FTL(0,1), lane), d0);
    v8f d1 = WMMA(a0, ld_bfrag(BhW, FTL(1,0), lane), cp1);
    d1     = WMMA(a1, ld_bfrag(BhW, FTL(1,1), lane), d1);

#pragma unroll
    for (int r = 0; r < 8; ++r) {
      float x0 = fmaf(rs, d0[r], bc0); x0 = (x0 > 0.f) ? x0 : 0.01f*x0;
      float x1 = fmaf(rs, d1[r], bc1); x1 = (x1 > 0.f) ? x1 : 0.01f*x1;
      int mm = r + grp*8;
      Dsw[mm*EMB + ln16]      = (h16)x0;
      Dsw[mm*EMB + 16 + ln16] = (h16)x1;
    }

    // ---- gates + LSTM cell ----
    v16h a_d  = ld_arow(Dsw + ln16*EMB, 0,  goff);
    v16h a_h0 = ld_arow(hrow,           0,  goff);
    v16h a_h1 = ld_arow(hrow,           32, goff);

#pragma unroll
    for (int jg = 0; jg < 4; ++jg) {
      v8f acc[4];
#pragma unroll
      for (int gt = 0; gt < 4; ++gt) {           // i, f, g, o
        int ut = gt*4 + jg;
        v8f z;
#pragma unroll
        for (int r = 0; r < 8; ++r) z[r] = 0.f;
        acc[gt] = WMMA(a_h0, ld_bfrag(BhW, FHH(ut,0), lane), z);
        acc[gt] = WMMA(a_h1, ld_bfrag(BhW, FHH(ut,1), lane), acc[gt]);
        acc[gt] = WMMA(a_d,  ld_bfrag(BhW, FIH(ut),   lane), acc[gt]);
      }
#pragma unroll
      for (int r = 0; r < 8; ++r) {
        float ii = fmaf(0.5f, tanh_fast(fmaf(acc[0][r], 0.5f, gbI[jg])), 0.5f);
        float ff = fmaf(0.5f, tanh_fast(fmaf(acc[1][r], 0.5f, gbF[jg])), 0.5f);
        float gg = tanh_fast(acc[2][r] + gbG[jg]);
        float oo = fmaf(0.5f, tanh_fast(fmaf(acc[3][r], 0.5f, gbO[jg])), 0.5f);
        float cn = ff*cf[jg][r] + ii*gg;
        cf[jg][r] = cn;
        Hsw[(r + grp*8)*HDIM + jg*16 + ln16] = (h16)(oo*tanh_fast(cn));
      }
    }

    // ---- LN2: stats from h fragments (lane + lane^16 hold complementary halves) ----
    v16h b_h0 = ld_arow(hrow, 0,  goff);
    v16h b_h1 = ld_arow(hrow, 32, goff);
    float s2 = 0.f, q2 = 0.f;
#pragma unroll
    for (int j = 0; j < 16; ++j) {
      float x = (float)b_h0[j]; s2 += x; q2 += x*x;
      float y = (float)b_h1[j]; s2 += y; q2 += y*y;
    }
    s2 += __shfl_xor(s2, 16, 32); q2 += __shfl_xor(q2, 16, 32);
    float mu2 = s2 * (1.f/64.f);
    float rs2 = rsqrtf(q2 * (1.f/64.f) - mu2*mu2 + 1e-5f);
    float nm2 = -mu2 * rs2;

    // normalize on the A side, then rel = a@B2 + Bc2
    v16h a2_0, a2_1;
#pragma unroll
    for (int j = 0; j < 16; ++j) {
      a2_0[j] = (h16)fmaf((float)b_h0[j], rs2, nm2);
      a2_1[j] = (h16)fmaf((float)b_h1[j], rs2, nm2);
    }
    v8f z2;
#pragma unroll
    for (int r = 0; r < 8; ++r) z2[r] = 0.f;
    v8f racc = WMMA(a2_0, ld_bfrag(BhW, FHP(0), lane), z2);
    racc     = WMMA(a2_1, ld_bfrag(BhW, FHP(1), lane), racc);

    int wslot = r0*16;   // physical tail slot written this step
#pragma unroll
    for (int r = 0; r < 8; ++r) {
      float rel = racc[r] + bc2;
      int mm = r + grp*8;
      if (ln16 < OUTD) {
        out[((size_t)t*NPED + pedBase + mm)*OUTD + ln16] = rel;
        TTw[mm*48 + wslot + ln16] = (h16)rel;
      }
    }
  }
}

extern "C" void kernel_launch(void* const* d_in, const int* in_sizes, int n_in,
                              void* d_out, int out_size, void* d_ws, size_t ws_size,
                              hipStream_t stream) {
  (void)in_sizes; (void)n_in; (void)out_size; (void)d_ws; (void)ws_size;
  const float* traj_rel = (const float*)d_in[1];
  const float* h0   = (const float*)d_in[2];
  const float* c0   = (const float*)d_in[3];
  const float* W_ih = (const float*)d_in[4];
  const float* W_hh = (const float*)d_in[5];
  const float* b_ih = (const float*)d_in[6];
  const float* b_hh = (const float*)d_in[7];
  const float* ln1g = (const float*)d_in[8];
  const float* ln1b = (const float*)d_in[9];
  const float* W_se = (const float*)d_in[10];
  const float* b_se = (const float*)d_in[11];
  const float* ln2g = (const float*)d_in[12];
  const float* ln2b = (const float*)d_in[13];
  const float* W_hp = (const float*)d_in[14];
  const float* b_hp = (const float*)d_in[15];

  dim3 grid(NPED / (NW*16));   // 512 workgroups, 16 peds per wave
  tdlstm_kernel<<<grid, TPB, SMEM_BYTES, stream>>>(
      traj_rel, h0, c0, W_ih, W_hh, b_ih, b_hh,
      ln1g, ln1b, W_se, b_se, ln2g, ln2b, W_hp, b_hp,
      (float*)d_out);
}
